// BandSplit_42253888258227
// MI455X (gfx1250) — compile-verified, matrix operand order checked by reading, not run on hardware
//
#include <hip/hip_runtime.h>
#include <hip/hip_bf16.h>
#include <stdint.h>

// Problem constants (from reference)
#define BB    4
#define CC    2
#define TT    1024
#define FF    1025     // NFFT/2+1
#define KB    64       // NBANDS
#define CO    128      // COUT
#define ROWS  (BB*TT)  // 4096 (b,t) rows
#define MTILE 64       // rows per block

typedef __attribute__((ext_vector_type(16))) __bf16 v16bf;
typedef __attribute__((ext_vector_type(8)))  float  v8f;

__device__ __forceinline__ unsigned short f2bf(float f) {
    unsigned int u = __float_as_uint(f);
    unsigned int r = u + 0x7FFFu + ((u >> 16) & 1u);   // round-to-nearest-even
    return (unsigned short)(r >> 16);
}

// Fragment swizzle (per CDNA5 ISA 16-bit operand layout):
// within a 32(K) x 16(row/col) block, element kk for row/col ml lives at
// lane = ml + 16*half, elem e, with half=(kk>>3)&1, e=(kk&7)|((kk>>4)<<3).
// A fragment uses ml = m; B fragment mirrors with ml = n. Each fragment block
// = 32 lanes * 16 bf16 = 1KB, so a fragment LOAD is one contiguous 32B read
// per lane (2x ds_load_b128).
__device__ __forceinline__ int frag_slot(int kk, int ml) {
    int h = (kk >> 3) & 1;
    int e = (kk & 7) | ((kk >> 4) << 3);
    return ((ml + (h << 4)) << 4) + e;
}

__device__ __forceinline__ v16bf load_frag(const unsigned short* fragBase, int block, int lane) {
    return *(const v16bf*)(fragBase + (block << 9) + (lane << 4));
}

__global__ __launch_bounds__(256)
void bandsplit_fused(const float* __restrict__ x,
                     const float* __restrict__ pre_w,
                     const float* __restrict__ pre_b,
                     const float* __restrict__ post_w,
                     const float* __restrict__ post_b,
                     const int*   __restrict__ idx,
                     const float* __restrict__ melw,
                     float*       __restrict__ out,
                     int W, int din, int DP)
{
    const int k       = blockIdx.y;          // band
    const int rowBase = blockIdx.x * MTILE;  // (b,t) row tile
    const int tid  = threadIdx.x;
    const int lane = tid & 31;
    const int wave = tid >> 5;
    const int KC1  = DP >> 5;                // K-chunks of GEMM1 (DP/32)
    const int NT2  = DP >> 4;                // N-tiles of GEMM2 (DP/16)

    extern __shared__ char smem[];
    const int aW = (W * 4 + 31) & ~31;
    int*            sIdx = (int*)smem;
    float*          sWt  = (float*)(smem + aW);
    // all fragment arrays: [tile][chunk] blocks of 512 ushorts (1KB)
    unsigned short* sGf  = (unsigned short*)(smem + 2 * aW); // 4 mtiles x KC1
    unsigned short* sZf  = sGf + 4 * KC1 * 512;              // 4 mtiles x 4 chunks
    unsigned short* sWf  = sZf + 4 * 4 * 512;                // KC1 x 8  /  4 x NT2

    const float* pw = pre_w  + (size_t)k * din * CO;
    const float* qw = post_w + (size_t)k * CO  * din;
    if (tid == 0) {            // warm per-band weight streams (global_prefetch_b8)
        __builtin_prefetch(pw, 0, 0);
        __builtin_prefetch(qw, 0, 0);
    }

    // ---- Phase A: band metadata + bf16 pre_w stage (fragment layout) ----
    for (int i = tid; i < W; i += 256) {
        sIdx[i] = idx[k * W + i];
        sWt[i]  = melw[k * W + i];
    }
    const int PWN = DP * CO;
    for (int e = tid; e < PWN; e += 256) {
        int i = e >> 7, o = e & (CO - 1);             // i: K(din) dim, o: N(CO) dim
        unsigned short v = f2bf(i < din ? pw[i * CO + o] : 0.f);
        int kc = i >> 5, kk = i & 31, nt = o >> 4, n = o & 15;
        sWf[((kc * 8 + nt) << 9) + frag_slot(kk, n)] = v;
    }
    __syncthreads();

    // ---- Phase B: gather G tile (64 x DP), melbank-weighted, bf16, A-frag layout ----
    for (int e = tid; e < MTILE * DP; e += 256) {
        int r = e / DP, j = e - r * DP;
        float v = 0.f;
        if (j < din) {
            int w = j >> 1, c = j & 1;
            int g = rowBase + r;
            int b = g >> 10, t = g & (TT - 1);
            v = x[(((size_t)(b * CC + c) * TT + t) * FF) + sIdx[w]] * sWt[w];
        }
        int mt = r >> 4, ml = r & 15, kc = j >> 5, kk = j & 31;
        sGf[((mt * KC1 + kc) << 9) + frag_slot(kk, ml)] = f2bf(v);
    }
    __syncthreads();

    // ---- GEMM1: Z(64x128) = G(64xDP) * pre_w(DPx128) + pre_b ----
    {
        const int mtile = wave >> 1;           // 0..3
        const int ntb   = (wave & 1) * 4;      // first of 4 n-tiles (cols 0-63 / 64-127)
        const float* pb = pre_b + k * CO;
        v8f acc[4];
#pragma unroll
        for (int nt = 0; nt < 4; ++nt) {
            float bv = pb[(ntb + nt) * 16 + (lane & 15)];
#pragma unroll
            for (int r = 0; r < 8; ++r) acc[nt][r] = bv;
        }
        for (int kc = 0; kc < KC1; ++kc) {
            v16bf a = load_frag(sGf, mtile * KC1 + kc, lane);
#pragma unroll
            for (int nt = 0; nt < 4; ++nt) {
                v16bf bm = load_frag(sWf, kc * 8 + ntb + nt, lane);
                acc[nt] = __builtin_amdgcn_wmma_f32_16x16x32_bf16(
                    false, a, false, bm, (short)0, acc[nt], false, false);
            }
        }
        // store Z directly in A-fragment layout for GEMM2 (K dim = CO)
#pragma unroll
        for (int nt = 0; nt < 4; ++nt) {
#pragma unroll
            for (int r = 0; r < 8; ++r) {
                int m = mtile * 16 + ((lane >> 4) << 3) + r;  // C layout row
                int n = (ntb + nt) * 16 + (lane & 15);        // C layout col
                int kc2 = n >> 5, kk2 = n & 31;
                sZf[((mtile * 4 + kc2) << 9) + frag_slot(kk2, m & 15)] =
                    f2bf(acc[nt][r]);
            }
        }
    }
    __syncthreads();

    // ---- stage bf16 post_w (CO x din) in B-fragment layout ----
    for (int e = tid; e < PWN; e += 256) {
        int d = e / DP, i = e - d * DP;               // d: K(CO) dim, i: N(din) dim
        unsigned short v = f2bf(i < din ? qw[d * din + i] : 0.f);
        int kc = d >> 5, kk = d & 31, nt = i >> 4, n = i & 15;
        sWf[((kc * NT2 + nt) << 9) + frag_slot(kk, n)] = v;
    }
    __syncthreads();

    // ---- GEMM2: Y(64xdin) = Z(64x128) * post_w(128xDP) + post_b; masked scatter-add ----
    {
        const int mtile = wave & 3;
        const float* qb = post_b + k * din;
        for (int nt = (wave >> 2); nt < NT2; nt += 2) {
            int j = nt * 16 + (lane & 15);
            float bv = (j < din) ? qb[j] : 0.f;
            v8f acc;
#pragma unroll
            for (int r = 0; r < 8; ++r) acc[r] = bv;
#pragma unroll
            for (int kc = 0; kc < 4; ++kc) {
                v16bf a  = load_frag(sZf, mtile * 4 + kc, lane);
                v16bf bm = load_frag(sWf, kc * NT2 + nt, lane);
                acc = __builtin_amdgcn_wmma_f32_16x16x32_bf16(
                    false, a, false, bm, (short)0, acc, false, false);
            }
            if (j < din) {
                int w = j >> 1, c = j & 1;
                if (sWt[w] != 0.f) {           // mask = (melw != 0)
                    int fi = sIdx[w];
#pragma unroll
                    for (int r = 0; r < 8; ++r) {
                        int m = mtile * 16 + ((lane >> 4) << 3) + r;
                        int g = rowBase + m;
                        int b = g >> 10, t = g & (TT - 1);
                        atomicAdd(out + (((size_t)(b * CC + c) * TT + t) * FF + fi),
                                  acc[r]);
                    }
                }
            }
        }
    }
}

__global__ __launch_bounds__(256)
void ola_scale(float* __restrict__ out, const float* __restrict__ ola, int n)
{
    int i = blockIdx.x * 256 + threadIdx.x;
    if (i < n) {
        int f = i % FF;
        out[i] = out[i] / ola[f];
    }
}

extern "C" void kernel_launch(void* const* d_in, const int* in_sizes, int n_in,
                              void* d_out, int out_size, void* d_ws, size_t ws_size,
                              hipStream_t stream)
{
    const float* x      = (const float*)d_in[0];
    const float* pre_w  = (const float*)d_in[1];
    const float* pre_b  = (const float*)d_in[2];
    const float* post_w = (const float*)d_in[3];
    const float* post_b = (const float*)d_in[4];
    const int*   idx    = (const int*)  d_in[5];
    const float* melw   = (const float*)d_in[6];
    const float* ola    = (const float*)d_in[8];
    float* out = (float*)d_out;

    const int W   = in_sizes[5] / KB;       // max band width
    const int din = 2 * W;
    const int DP  = (din + 31) & ~31;       // pad K dim to WMMA chunk
    const int KC1 = DP >> 5;
    const int aW  = (W * 4 + 31) & ~31;

    size_t lds = (size_t)2 * aW
               + (size_t)4 * KC1 * 512 * 2  // sGf  (== 64*DP bf16)
               + (size_t)16 * 512 * 2       // sZf  (== 64*128 bf16)
               + (size_t)KC1 * 8 * 512 * 2; // sWf  (== DP*128 bf16)
    (void)hipFuncSetAttribute((const void*)bandsplit_fused,
                              hipFuncAttributeMaxDynamicSharedMemorySize, (int)lds);

    (void)hipMemsetAsync(d_out, 0, (size_t)out_size * sizeof(float), stream);

    dim3 grid(ROWS / MTILE, KB);            // 64 row tiles x 64 bands
    bandsplit_fused<<<grid, 256, lds, stream>>>(x, pre_w, pre_b, post_w, post_b,
                                                idx, melw, out, W, din, DP);

    int n = out_size;
    ola_scale<<<(n + 255) / 256, 256, 0, stream>>>(out, ola, n);
}
